// att_trans_24489903522165
// MI455X (gfx1250) — compile-verified, matrix operand order checked by reading, not run on hardware
//
#include <hip/hip_runtime.h>
#include <hip/hip_bf16.h>
#include <math.h>

// ---------------------------------------------------------------------------
// MI455X / gfx1250: wave32, WMMA bf16 16x16x32 f32-acc, async global->LDS DMA
// ---------------------------------------------------------------------------

typedef __bf16 bf16_t;
typedef __attribute__((ext_vector_type(16))) __bf16 v16bf;
typedef __attribute__((ext_vector_type(8)))  __bf16 v8bf;
typedef __attribute__((ext_vector_type(8)))  float  v8f;

#define B_ 8
#define S_ 2048
#define E_ 512

// Async copy 16B global -> LDS (GLOBAL_LOAD_ASYNC_TO_LDS_B128, GV mode,
// tracked by ASYNCcnt).  lds_off = raw LDS byte offset, g = global address.
__device__ __forceinline__ void async_cp_b128(unsigned lds_off, const void* g) {
  asm volatile("global_load_async_to_lds_b128 %0, %1, off"
               :
               : "v"(lds_off), "v"((unsigned long long)(size_t)g)
               : "memory");
}
__device__ __forceinline__ void wait_async0() {
  asm volatile("s_wait_asynccnt 0x0" ::: "memory");
}
__device__ __forceinline__ unsigned lds_off(const void* p) {
  return (unsigned)(size_t)p;        // generic LDS pointer -> addr[31:0]
}

// 16x32 bf16 fragment (A-layout; B-frags are built from B^T rows -> identical
// per-lane layout). ISA 7.12.2: lanes 0-15: row=lane, K={0..7,16..23};
// lanes 16-31: row=lane-16, K={8..15,24..31}.
__device__ __forceinline__ v16bf load_frag(const bf16_t* tile, int ld, int lane) {
  int r  = lane & 15;
  int kh = (lane >> 4) << 3;
  const bf16_t* p = tile + (long long)r * ld + kh;
  v8bf lo = *(const v8bf*)(p);
  v8bf hi = *(const v8bf*)(p + 16);
  return __builtin_shufflevector(lo, hi, 0,1,2,3,4,5,6,7,
                                         8,9,10,11,12,13,14,15);
}

// ---------------------------------------------------------------------------
// xp = x + sinusoidal positional encoding; emit f32 (residual) + bf16 (GEMM A)
// ---------------------------------------------------------------------------
__global__ void k_pe_xp(const float* __restrict__ x,
                        float* __restrict__ xp_f, bf16_t* __restrict__ xp_b) {
  long long i = (long long)blockIdx.x * blockDim.x + threadIdx.x;  // < B*S*E
  int e = (int)(i % E_);
  int s = (int)((i / E_) % S_);
  float dv  = __expf(-(float)(e & ~1) * (9.21034037197618f / (float)E_));
  float ang = (float)s * dv;
  float pe  = (e & 1) ? __cosf(ang) : __sinf(ang);
  float v = x[i] + pe;
  xp_f[i] = v;
  xp_b[i] = (bf16_t)v;
}

// ---------------------------------------------------------------------------
// tc[b][n] = bq[n] + sum_k t[b,k]*Wq[E+k][n] + c[b,k]*Wq[2E+k][n]
// ---------------------------------------------------------------------------
__global__ void k_tc(const float* __restrict__ t, const float* __restrict__ c,
                     const float* __restrict__ Wq, const float* __restrict__ bq,
                     float* __restrict__ tc) {
  int i = blockIdx.x * blockDim.x + threadIdx.x;   // < B*E
  int n = i & (E_ - 1);
  int b = i >> 9;
  float acc = bq[n];
  const float* tb = t + b * E_;
  const float* cb = c + b * E_;
  for (int k = 0; k < E_; ++k) {
    acc += tb[k] * Wq[(long long)(E_     + k) * E_ + n];
    acc += cb[k] * Wq[(long long)(2 * E_ + k) * E_ + n];
  }
  tc[i] = acc;
}

// dst_bf16[n*K + k] = (bf16) src_f32[k*N + n]   (weight transpose+convert)
__global__ void k_transcvt(const float* __restrict__ src,
                           bf16_t* __restrict__ dst, int K, int N) {
  long long i = (long long)blockIdx.x * blockDim.x + threadIdx.x;  // < K*N
  int n = (int)(i % N);
  int k = (int)(i / N);
  dst[(long long)n * K + k] = (bf16_t)src[i];
}

// qT[b][e][s] = q[b][s][e]  (bf16)
__global__ void k_qtrans(const bf16_t* __restrict__ q, bf16_t* __restrict__ qT) {
  long long i = (long long)blockIdx.x * blockDim.x + threadIdx.x;  // < B*S*E
  int e = (int)(i % E_);
  long long se = i / E_;
  int s = (int)(se % S_);
  int b = (int)(se / S_);
  qT[((long long)b * E_ + e) * S_ + s] = q[i];
}

// ---------------------------------------------------------------------------
// Batched GEMM, async double-buffered LDS staging.
//   C[b] = A[b] (MxK row-major bf16) * B (KxN), B supplied as BT (NxK bf16).
// Block = 256 thr = 8 waves (2M x 4N), wave tile 32M x 64N (2 m-tiles share
// the 4 B-frags).  Workgroup tile 64M x 256N, K-stage 64.
// LDS/stage: A 64x64 (8KB) + BT 256x64 (32KB); two buffers = 80KB.
// Pipeline: issue async(s+1) -> frag-preload + 8 back-to-back WMMAs per
// k-step on stage s -> s_wait_asynccnt 0 -> barrier.
// ---------------------------------------------------------------------------
#define KSTG   64
#define ATILE  (64 * KSTG)            // 4096 elems
#define BTILE  (256 * KSTG)           // 16384 elems
#define STAGE  (ATILE + BTILE)        // 20480 elems (40,960 B)

__global__ void __launch_bounds__(256)
k_gemm(const bf16_t* __restrict__ A, const bf16_t* __restrict__ BT,
       bf16_t* __restrict__ C, int K, int N,
       long long sA, long long sB, long long sC,
       const float* __restrict__ bias, long long sBias,
       float scale, int relu) {
  __shared__ __align__(16) bf16_t smem[2 * STAGE];

  int tid  = threadIdx.x;
  int lane = tid & 31;
  int wid  = tid >> 5;
  int wm = wid >> 2, wn = wid & 3;
  int rowblk = blockIdx.x * 64;
  int colblk = blockIdx.y * 256;
  int b = blockIdx.z;

  const bf16_t* Ag  = A  + (long long)b * sA + (long long)rowblk * K;
  const bf16_t* BTg = BT + (long long)b * sB + (long long)colblk * K;
  C += (long long)b * sC;

  // prefetch addressing: A 64x64 -> 2 chunks/thread; BT 256x64 -> 8/thread
  int arow = tid >> 2, acol = (tid & 3) * 16;
  const bf16_t* agp = Ag + (long long)arow * K + acol;
  const bf16_t* bgp = BTg + (long long)tid * K;
  unsigned aoff = lds_off(smem + arow * KSTG + acol);
  unsigned boff = lds_off(smem + ATILE + tid * KSTG);

  int nstages = K / KSTG;

  // stage 0 prefetch
  async_cp_b128(aoff, agp);
  async_cp_b128(aoff + 16, agp + 8);
#pragma unroll
  for (int j = 0; j < 8; ++j) async_cp_b128(boff + j * 16, bgp + j * 8);
  wait_async0();
  __syncthreads();

  v8f zero = {0.f,0.f,0.f,0.f,0.f,0.f,0.f,0.f};
  v8f acc[2][4];
#pragma unroll
  for (int mt = 0; mt < 2; ++mt)
#pragma unroll
    for (int nt = 0; nt < 4; ++nt) acc[mt][nt] = zero;

  for (int s = 0; s < nstages; ++s) {
    int cur = s & 1;
    if (s + 1 < nstages) {                     // overlap DMA with WMMAs
      int k0 = (s + 1) * KSTG;
      unsigned bufo = (cur ^ 1) * (STAGE * 2); // other buffer, byte offset
      async_cp_b128(aoff + bufo, agp + k0);
      async_cp_b128(aoff + bufo + 16, agp + k0 + 8);
      const bf16_t* bg = bgp + k0;
#pragma unroll
      for (int j = 0; j < 8; ++j) async_cp_b128(boff + bufo + j * 16, bg + j * 8);
    }
    const bf16_t* sAc = smem + cur * STAGE;
    const bf16_t* sBc = sAc + ATILE;
#pragma unroll
    for (int ks = 0; ks < 2; ++ks) {
      // preload all frags for this k-step, then 8 WMMAs back-to-back
      v16bf a0 = load_frag(sAc + (wm * 32 +  0) * KSTG + ks * 32, KSTG, lane);
      v16bf a1 = load_frag(sAc + (wm * 32 + 16) * KSTG + ks * 32, KSTG, lane);
      v16bf bfr[4];
#pragma unroll
      for (int nt = 0; nt < 4; ++nt)
        bfr[nt] = load_frag(sBc + (wn * 64 + nt * 16) * KSTG + ks * 32, KSTG, lane);
#pragma unroll
      for (int nt = 0; nt < 4; ++nt)
        acc[0][nt] = __builtin_amdgcn_wmma_f32_16x16x32_bf16(
            false, a0, false, bfr[nt], (short)0, acc[0][nt], false, false);
#pragma unroll
      for (int nt = 0; nt < 4; ++nt)
        acc[1][nt] = __builtin_amdgcn_wmma_f32_16x16x32_bf16(
            false, a1, false, bfr[nt], (short)0, acc[1][nt], false, false);
    }
    if (s + 1 < nstages) {
      wait_async0();       // our stage-(s+1) DMA landed
      __syncthreads();     // all waves' DMA visible; buffers safe to swap
    }
  }

  // C/D layout: lanes 0-15 -> M=r, lanes 16-31 -> M=r+8; N = lane&15
  int mr = (lane >> 4) << 3;
  int nc = lane & 15;
#pragma unroll
  for (int mt = 0; mt < 2; ++mt) {
    int row0 = rowblk + wm * 32 + mt * 16;
#pragma unroll
    for (int nt = 0; nt < 4; ++nt) {
      int col = colblk + wn * 64 + nt * 16 + nc;
      float bv = bias ? bias[(long long)b * sBias + col] : 0.0f;
#pragma unroll
      for (int r = 0; r < 8; ++r) {
        float v = acc[mt][nt][r] * scale + bv;
        if (relu) v = fmaxf(v, 0.0f);
        C[(long long)(row0 + mr + r) * N + col] = (bf16_t)v;
      }
    }
  }
}

// ---------------------------------------------------------------------------
// Fused: h = hmid @ W2 + b2 ; LayerNorm(h)*gamma+beta + xp  -> out (f32)
// One block = 16 rows x full 512 cols (8 waves, 4 n-tiles each).
// ---------------------------------------------------------------------------
__global__ void __launch_bounds__(256)
k_mlp2_ln(const bf16_t* __restrict__ Hm, const bf16_t* __restrict__ W2T,
          const float* __restrict__ b2, const float* __restrict__ gamma,
          const float* __restrict__ beta, const float* __restrict__ xp,
          float* __restrict__ out) {
  __shared__ float sh[16][E_];
  __shared__ float ps[16][16];
  __shared__ float pq[16][16];
  __shared__ float smu[16];
  __shared__ float srs[16];

  int lane = threadIdx.x & 31;
  int wid  = threadIdx.x >> 5;
  int b    = blockIdx.z;
  int row0 = blockIdx.x * 16;
  int col0 = wid * 64;

  const bf16_t* Ab = Hm + ((long long)b * S_ + row0) * E_;

  v8f zero = {0.f,0.f,0.f,0.f,0.f,0.f,0.f,0.f};
  v8f acc[4];
#pragma unroll
  for (int i = 0; i < 4; ++i) acc[i] = zero;

  for (int k = 0; k < E_; k += 32) {
    // preload frags, then 4 WMMAs back-to-back
    v16bf a = load_frag(Ab + k, E_, lane);
    v16bf w[4];
#pragma unroll
    for (int nt = 0; nt < 4; ++nt)
      w[nt] = load_frag(W2T + (long long)(col0 + nt * 16) * E_ + k, E_, lane);
#pragma unroll
    for (int nt = 0; nt < 4; ++nt)
      acc[nt] = __builtin_amdgcn_wmma_f32_16x16x32_bf16(
          false, a, false, w[nt], (short)0, acc[nt], false, false);
  }

  int mr = (lane >> 4) << 3;
  int nc = lane & 15;
#pragma unroll
  for (int nt = 0; nt < 4; ++nt) {
    int col = col0 + nt * 16 + nc;
    float bv = b2[col];
#pragma unroll
    for (int r = 0; r < 8; ++r) sh[mr + r][col] = acc[nt][r] + bv;
  }
  __syncthreads();

  {
    int row = threadIdx.x >> 4;
    int seg = threadIdx.x & 15;
    float s = 0.f, q = 0.f;
#pragma unroll 4
    for (int i = 0; i < 32; ++i) {
      float h = sh[row][seg * 32 + i];
      s += h; q += h * h;
    }
    ps[row][seg] = s;
    pq[row][seg] = q;
  }
  __syncthreads();
  if (threadIdx.x < 16) {
    float s = 0.f, q = 0.f;
    for (int i = 0; i < 16; ++i) { s += ps[threadIdx.x][i]; q += pq[threadIdx.x][i]; }
    float mu  = s * (1.0f / E_);
    float var = q * (1.0f / E_) - mu * mu;
    smu[threadIdx.x] = mu;
    srs[threadIdx.x] = rsqrtf(var + 1e-5f);
  }
  __syncthreads();

  const float* xpb = xp  + ((long long)b * S_ + row0) * E_;
  float*       ob  = out + ((long long)b * S_ + row0) * E_;
#pragma unroll 4
  for (int i = 0; i < 32; ++i) {
    int idx = threadIdx.x + i * 256;     // < 16*512
    int r = idx >> 9;
    int c2 = idx & 511;
    float v = (sh[r][c2] - smu[r]) * srs[r] * gamma[c2] + beta[c2];
    ob[(long long)r * E_ + c2] = v + xpb[(long long)r * E_ + c2];
  }
}

// ---------------------------------------------------------------------------
// Host launch
// ---------------------------------------------------------------------------
extern "C" void kernel_launch(void* const* d_in, const int* in_sizes, int n_in,
                              void* d_out, int out_size, void* d_ws, size_t ws_size,
                              hipStream_t stream) {
  const float* x     = (const float*)d_in[0];
  const float* t     = (const float*)d_in[1];
  const float* c     = (const float*)d_in[2];
  const float* Wq    = (const float*)d_in[3];
  const float* bq    = (const float*)d_in[4];
  const float* W1    = (const float*)d_in[5];
  const float* b1    = (const float*)d_in[6];
  const float* W2    = (const float*)d_in[7];
  const float* b2    = (const float*)d_in[8];
  const float* gamma = (const float*)d_in[9];
  const float* beta  = (const float*)d_in[10];
  float* out = (float*)d_out;

  char* w = (char*)d_ws;
  float*  xp_f  = (float*) (w + 0);            //  33,554,432 B
  bf16_t* xp_b  = (bf16_t*)(w + 33554432);     //  16,777,216
  bf16_t* q_b   = (bf16_t*)(w + 50331648);     //  16,777,216
  bf16_t* qT_b  = (bf16_t*)(w + 67108864);     //  16,777,216
  bf16_t* att_b = (bf16_t*)(w + 83886080);     //  67,108,864
  bf16_t* v2_b  = (bf16_t*)(w + 150994944);    //  16,777,216
  bf16_t* hm_b  = (bf16_t*)(w + 167772160);    //  16,777,216
  bf16_t* WqT   = (bf16_t*)(w + 184549376);    //     524,288
  bf16_t* W1T   = (bf16_t*)(w + 185073664);    //     524,288
  bf16_t* W2T   = (bf16_t*)(w + 185597952);    //     524,288
  float*  tc    = (float*) (w + 186122240);    //      16,384

  const long long SE = (long long)S_ * E_;
  const long long SS = (long long)S_ * S_;
  dim3 blk(256);

  k_pe_xp<<<(B_ * S_ * E_) / 256, blk, 0, stream>>>(x, xp_f, xp_b);
  k_tc<<<(B_ * E_) / 256, blk, 0, stream>>>(t, c, Wq, bq, tc);
  k_transcvt<<<(E_ * E_) / 256, blk, 0, stream>>>(Wq, WqT, E_, E_);  // top E rows
  k_transcvt<<<(E_ * E_) / 256, blk, 0, stream>>>(W1, W1T, E_, E_);
  k_transcvt<<<(E_ * E_) / 256, blk, 0, stream>>>(W2, W2T, E_, E_);

  // q = xp @ Wq_top + tc[b]          (M=S, N=E, K=E)
  k_gemm<<<dim3(S_ / 64, E_ / 256, B_), blk, 0, stream>>>(
      xp_b, WqT, q_b, E_, E_, SE, 0, SE, tc, E_, 1.0f, 0);

  k_qtrans<<<(B_ * S_ * E_) / 256, blk, 0, stream>>>(q_b, qT_b);

  // att = (q @ q^T) / E              (M=S, N=S, K=E); BT operand == q itself
  k_gemm<<<dim3(S_ / 64, S_ / 256, B_), blk, 0, stream>>>(
      q_b, q_b, att_b, E_, S_, SE, SE, SS, nullptr, 0, 1.0f / (float)E_, 0);

  // v2 = att @ q                     (M=S, N=E, K=S); BT operand == qT
  k_gemm<<<dim3(S_ / 64, E_ / 256, B_), blk, 0, stream>>>(
      att_b, qT_b, v2_b, S_, E_, SS, (long long)E_ * S_, SE, nullptr, 0, 1.0f, 0);

  // hmid = relu(v2 @ W1 + b1)        (M=S, N=E, K=E)
  k_gemm<<<dim3(S_ / 64, E_ / 256, B_), blk, 0, stream>>>(
      v2_b, W1T, hm_b, E_, E_, SE, 0, SE, b1, 0, 1.0f, 1);

  // h = hmid @ W2 + b2 ; LN ; +xp    (fused)
  k_mlp2_ln<<<dim3(S_ / 16, 1, B_), blk, 0, stream>>>(
      hm_b, W2T, b2, gamma, beta, xp_f, out);
}
